// AttentionDecoderRNN_7550552507258
// MI455X (gfx1250) — matrix operand order repaired and checked by reading, compiled, not run on hardware
//
#include <hip/hip_runtime.h>
#include <stdint.h>

// ---------------------------------------------------------------------------
// Problem constants
// ---------------------------------------------------------------------------
#define Hdim 1024
#define Bdim 256
#define Sdim 128
#define VOCdim 5292

typedef __bf16 bf16_t;
typedef __attribute__((ext_vector_type(16))) __bf16 v16bf;
typedef __attribute__((ext_vector_type(8)))  float  v8f;

// WMMA macro-tile: workgroup = 128(M) x 128(N), K-block 64, 8 waves (4x2),
// wave tile = 32(M) x 64(N) -> 16 WMMAs per K-block per wave.
#define BM 128
#define BN 128
#define BK 64
#define LDSP 72           // LDS row pitch in bf16 (144B: 16B-aligned, bank-friendly)
#define TILE_E (128 * LDSP)

// ---------------------------------------------------------------------------
// gfx1250 fast-path helpers (guarded; host pass / other toolchains fall back)
// ---------------------------------------------------------------------------
#if defined(__AMDGCN__) && __has_builtin(__builtin_amdgcn_global_load_async_to_lds_b128)
#define HAVE_ASYNC_LDS 1
#else
#define HAVE_ASYNC_LDS 0
#endif

__device__ inline void async_copy_b128(bf16_t* lds_dst, const bf16_t* gsrc) {
#if HAVE_ASYNC_LDS
  typedef int v4i __attribute__((ext_vector_type(4)));
  typedef __attribute__((address_space(1))) v4i* gv4i_p;
  typedef __attribute__((address_space(3))) v4i* lv4i_p;
  const __attribute__((address_space(1))) void* gp =
      (const __attribute__((address_space(1))) void*)gsrc;
  __attribute__((address_space(3))) void* lp =
      (__attribute__((address_space(3))) void*)lds_dst;
  __builtin_amdgcn_global_load_async_to_lds_b128((gv4i_p)gp, (lv4i_p)lp, 0, 0);
#else
  *(uint4*)lds_dst = *(const uint4*)gsrc;
#endif
}

__device__ inline void wait_async_all() {
#if HAVE_ASYNC_LDS
#if __has_builtin(__builtin_amdgcn_s_wait_asynccnt)
  __builtin_amdgcn_s_wait_asynccnt(0);
#else
  asm volatile("s_wait_asynccnt 0" ::: "memory");
#endif
#endif
}

__device__ inline float fast_tanh(float x) {
#if defined(__AMDGCN__) && __has_builtin(__builtin_amdgcn_tanhf)
  return __builtin_amdgcn_tanhf(x);           // v_tanh_f32 (CDNA5 TRANS op)
#else
  return tanhf(x);
#endif
}

__device__ inline float fast_exp(float x) {   // e^x via v_exp_f32 (2^x)
#if defined(__AMDGCN__)
  return __builtin_amdgcn_exp2f(x * 1.4426950408889634f);
#else
  return expf(x);
#endif
}

__device__ inline float fast_log(float x) {   // ln(x) via v_log_f32 (log2)
#if defined(__AMDGCN__)
  return __builtin_amdgcn_logf(x) * 0.6931471805599453f;
#else
  return logf(x);
#endif
}

__device__ inline float fast_sigmoid(float x) {
#if defined(__AMDGCN__)
  float e = __builtin_amdgcn_exp2f(-1.4426950408889634f * x);
  return __builtin_amdgcn_rcpf(1.0f + e);
#else
  return 1.f / (1.f + expf(-x));
#endif
}

// ---------------------------------------------------------------------------
// Fragment loaders (CDNA5 ISA 7.12.2 layouts)
// A 16x32 bf16: lanes 0-15 -> M=lane, K {k..k+7, k+16..k+23};
//               lanes 16-31 -> M=lane-16, K {k+8..k+15, k+24..k+31}
// ---------------------------------------------------------------------------
__device__ inline v16bf load_fragA(const bf16_t* tile, int row0, int kofs) {
  int lane = threadIdx.x & 31;
  int m  = (lane & 15) + row0;
  int k0 = ((lane < 16) ? 0 : 8) + kofs;
  const bf16_t* p = tile + m * LDSP + k0;
  union { uint4 u[2]; v16bf v; } t;
  t.u[0] = *(const uint4*)(p);
  t.u[1] = *(const uint4*)(p + 16);
  return t.v;
}

// B 32x16 bf16 (from Bt tile: N rows x K cols): lanes 0-15 -> K {k..k+7, k+16..k+23},
// lanes 16-31 -> K {k+8..k+15, k+24..k+31}, N = lane&15
__device__ inline v16bf load_fragB(const bf16_t* tile, int row0, int kofs) {
  int lane = threadIdx.x & 31;
  int n  = (lane & 15) + row0;
  int kh = ((lane < 16) ? 0 : 8) + kofs;
  const bf16_t* p = tile + n * LDSP + kh;
  union { uint4 u[2]; v16bf v; } t;
  t.u[0] = *(const uint4*)(p);
  t.u[1] = *(const uint4*)(p + 16);
  return t.v;
}

#define WMMA_BF16(a, b, c) \
  __builtin_amdgcn_wmma_f32_16x16x32_bf16(false, (a), false, (b), (short)0, (c), false, false)

// ---------------------------------------------------------------------------
// Small utility kernels
// ---------------------------------------------------------------------------
__global__ void f32_to_bf16_kernel(const float* __restrict__ in,
                                   bf16_t* __restrict__ out, size_t n) {
  size_t i = (size_t)blockIdx.x * blockDim.x + threadIdx.x;
  size_t stride = (size_t)gridDim.x * blockDim.x;
  for (; i < n; i += stride) out[i] = (bf16_t)in[i];
}

__global__ void pack_hidden_kernel(const float* __restrict__ hidden,
                                   bf16_t* __restrict__ xbuf) {
  int idx = blockIdx.x * blockDim.x + threadIdx.x;   // B*H
  int b = idx >> 10, h = idx & (Hdim - 1);
  xbuf[(size_t)b * (3 * Hdim) + h] = (bf16_t)hidden[idx];
}

__global__ void zero_kernel(float* __restrict__ p, size_t n) {
  size_t i = (size_t)blockIdx.x * blockDim.x + threadIdx.x;
  size_t stride = (size_t)gridDim.x * blockDim.x;
  for (; i < n; i += stride) p[i] = 0.f;
}

// ---------------------------------------------------------------------------
// Fused attention-energy GEMM (software-pipelined, double-buffered LDS):
// enc[S*B,2H](f32) x attn1_w[:,H:].T(bf16) -> tanh(+hproj+b1)*w2 -> atomicAdd
// ---------------------------------------------------------------------------
__global__ __launch_bounds__(256) void attn_energy_kernel(
    const float*  __restrict__ enc,      // [S*B, 2H]
    const bf16_t* __restrict__ Bt,       // attn1w_bf + H ; ldb = 3H
    const float*  __restrict__ hproj,    // [B, H]
    const float*  __restrict__ attn1_b,  // [H]
    const float*  __restrict__ attn2_w,  // [H]
    float*        __restrict__ energies) // [S*B] pre-zeroed
{
  __shared__ alignas(16) bf16_t sA[2][TILE_E];
  __shared__ alignas(16) bf16_t sB[2][TILE_E];
  const int tid  = threadIdx.x;
  const int lane = tid & 31;
  const int wave = tid >> 5;
  const int wm = wave & 3;         // 4 waves along M
  const int wn = wave >> 2;        // 2 waves along N
  const int rowBase = blockIdx.x * BM;
  const int colBase = blockIdx.y * BN;
  const int lda = 2 * Hdim;
  const int ldb = 3 * Hdim;
  const int K   = 2 * Hdim;

  // staging assignment: 2 threads per row, 32 elements each
  const int rA = tid >> 1, cA = (tid & 1) * 32;           // A: f32 elements
  const int nB = tid >> 1, kB = (tid & 1) * 32;           // Bt: bf16 elements
  const float*  gA = enc + (size_t)(rowBase + rA) * lda + cA;
  const bf16_t* gB = Bt  + (size_t)(colBase + nB) * ldb + kB;
  bf16_t* dA = &sA[0][0] + rA * LDSP + cA;
  bf16_t* dB = &sB[0][0] + nB * LDSP + kB;

  float4 areg[8];
  v8f acc[2][4] = {};

  // ---- prologue: block 0
#pragma unroll
  for (int i = 0; i < 8; ++i) areg[i] = ((const float4*)gA)[i];
#pragma unroll
  for (int i = 0; i < 4; ++i) async_copy_b128(dB + 8 * i, gB + 8 * i);
  {
    union { bf16_t h[32]; uint4 u[4]; } tmp;
#pragma unroll
    for (int i = 0; i < 8; ++i) {
      tmp.h[4*i+0] = (bf16_t)areg[i].x; tmp.h[4*i+1] = (bf16_t)areg[i].y;
      tmp.h[4*i+2] = (bf16_t)areg[i].z; tmp.h[4*i+3] = (bf16_t)areg[i].w;
    }
#pragma unroll
    for (int i = 0; i < 4; ++i) *(uint4*)(dA + 8 * i) = tmp.u[i];
  }
  wait_async_all();
  __syncthreads();

  for (int kb = 0, buf = 0; kb < K; kb += BK, buf ^= 1) {
    const bf16_t* curA = &sA[buf][0];
    const bf16_t* curB = &sB[buf][0];
    bf16_t* nxtA = &sA[buf ^ 1][0] + rA * LDSP + cA;
    bf16_t* nxtB = &sB[buf ^ 1][0] + nB * LDSP + kB;
    const bool has_next = (kb + BK) < K;
    if (has_next) {       // issue next-block loads before compute (overlap)
#pragma unroll
      for (int i = 0; i < 8; ++i) areg[i] = ((const float4*)(gA + kb + BK))[i];
#pragma unroll
      for (int i = 0; i < 4; ++i) async_copy_b128(nxtB + 8 * i, gB + kb + BK + 8 * i);
    }
#pragma unroll
    for (int ks = 0; ks < BK; ks += 32) {
      v16bf a0 = load_fragA(curA, wm * 32,      ks);
      v16bf a1 = load_fragA(curA, wm * 32 + 16, ks);
      v16bf b0 = load_fragB(curB, wn * 64,      ks);
      v16bf b1 = load_fragB(curB, wn * 64 + 16, ks);
      v16bf b2 = load_fragB(curB, wn * 64 + 32, ks);
      v16bf b3 = load_fragB(curB, wn * 64 + 48, ks);
      acc[0][0] = WMMA_BF16(a0, b0, acc[0][0]);
      acc[0][1] = WMMA_BF16(a0, b1, acc[0][1]);
      acc[0][2] = WMMA_BF16(a0, b2, acc[0][2]);
      acc[0][3] = WMMA_BF16(a0, b3, acc[0][3]);
      acc[1][0] = WMMA_BF16(a1, b0, acc[1][0]);
      acc[1][1] = WMMA_BF16(a1, b1, acc[1][1]);
      acc[1][2] = WMMA_BF16(a1, b2, acc[1][2]);
      acc[1][3] = WMMA_BF16(a1, b3, acc[1][3]);
    }
    if (has_next) {       // convert+store A (load waits land here), drain DMA
      union { bf16_t h[32]; uint4 u[4]; } tmp;
#pragma unroll
      for (int i = 0; i < 8; ++i) {
        tmp.h[4*i+0] = (bf16_t)areg[i].x; tmp.h[4*i+1] = (bf16_t)areg[i].y;
        tmp.h[4*i+2] = (bf16_t)areg[i].z; tmp.h[4*i+3] = (bf16_t)areg[i].w;
      }
#pragma unroll
      for (int i = 0; i < 4; ++i) *(uint4*)(nxtA + 8 * i) = tmp.u[i];
      wait_async_all();
    }
    __syncthreads();
  }

  // Epilogue: e[row] += sum_j tanh(acc + attn1_b[j] + hproj[b,j]) * attn2_w[j]
  const int n_lane = lane & 15;
  const int m_hi   = (lane >> 4) * 8;
  float bia[4], w2[4];
  int   jcol[4];
#pragma unroll
  for (int nf = 0; nf < 4; ++nf) {
    jcol[nf] = colBase + wn * 64 + nf * 16 + n_lane;
    bia[nf]  = attn1_b[jcol[nf]];
    w2[nf]   = attn2_w[jcol[nf]];
  }
#pragma unroll
  for (int mf = 0; mf < 2; ++mf) {
#pragma unroll
    for (int r = 0; r < 8; ++r) {
      int row = rowBase + wm * 32 + mf * 16 + m_hi + r;
      int b   = row & (Bdim - 1);
      float sum = 0.f;
#pragma unroll
      for (int nf = 0; nf < 4; ++nf) {
        float v = acc[mf][nf][r] + bia[nf] + hproj[(size_t)b * Hdim + jcol[nf]];
        sum += fast_tanh(v) * w2[nf];
      }
      sum += __shfl_xor(sum, 1, 32);
      sum += __shfl_xor(sum, 2, 32);
      sum += __shfl_xor(sum, 4, 32);
      sum += __shfl_xor(sum, 8, 32);
      if (n_lane == 0) atomicAdd(&energies[row], sum);
    }
  }
}

// ---------------------------------------------------------------------------
// Generic bf16 WMMA GEMM: C[M,N](f32) = A[M,K] @ Bt[N,K]^T (+bias0+bias1 | +=)
// Async double-buffered LDS staging for both operands.
// ---------------------------------------------------------------------------
__global__ __launch_bounds__(256) void gemm_bt_kernel(
    const bf16_t* __restrict__ A,  int lda,
    const bf16_t* __restrict__ Bt, int ldb,
    float*        __restrict__ C,  int ldc,
    int M, int N, int K,
    const float* __restrict__ bias0,
    const float* __restrict__ bias1,
    int accum)
{
  __shared__ alignas(16) bf16_t sA[2][TILE_E];
  __shared__ alignas(16) bf16_t sB[2][TILE_E];
  const int tid  = threadIdx.x;
  const int lane = tid & 31;
  const int wave = tid >> 5;
  const int wm = wave & 3;
  const int wn = wave >> 2;
  const int rowBase = blockIdx.x * BM;
  const int colBase = blockIdx.y * BN;

  const int rA = tid >> 1, cA = (tid & 1) * 32;
  const int nB = tid >> 1, kB = (tid & 1) * 32;
  int gr = rowBase + rA; if (gr >= M) gr = M - 1;
  int gn = colBase + nB; if (gn >= N) gn = N - 1;
  const bf16_t* gA = A  + (size_t)gr * lda + cA;
  const bf16_t* gB = Bt + (size_t)gn * ldb + kB;
  bf16_t* dA = &sA[0][0] + rA * LDSP + cA;
  bf16_t* dB = &sB[0][0] + nB * LDSP + kB;

  v8f acc[2][4] = {};

  // prologue: block 0
#pragma unroll
  for (int i = 0; i < 4; ++i) async_copy_b128(dA + 8 * i, gA + 8 * i);
#pragma unroll
  for (int i = 0; i < 4; ++i) async_copy_b128(dB + 8 * i, gB + 8 * i);
  wait_async_all();
  __syncthreads();

  for (int kb = 0, buf = 0; kb < K; kb += BK, buf ^= 1) {
    const bf16_t* curA = &sA[buf][0];
    const bf16_t* curB = &sB[buf][0];
    bf16_t* nxtA = &sA[buf ^ 1][0] + rA * LDSP + cA;
    bf16_t* nxtB = &sB[buf ^ 1][0] + nB * LDSP + kB;
    const bool has_next = (kb + BK) < K;
    if (has_next) {
#pragma unroll
      for (int i = 0; i < 4; ++i) async_copy_b128(nxtA + 8 * i, gA + kb + BK + 8 * i);
#pragma unroll
      for (int i = 0; i < 4; ++i) async_copy_b128(nxtB + 8 * i, gB + kb + BK + 8 * i);
    }
#pragma unroll
    for (int ks = 0; ks < BK; ks += 32) {
      v16bf a0 = load_fragA(curA, wm * 32,      ks);
      v16bf a1 = load_fragA(curA, wm * 32 + 16, ks);
      v16bf b0 = load_fragB(curB, wn * 64,      ks);
      v16bf b1 = load_fragB(curB, wn * 64 + 16, ks);
      v16bf b2 = load_fragB(curB, wn * 64 + 32, ks);
      v16bf b3 = load_fragB(curB, wn * 64 + 48, ks);
      acc[0][0] = WMMA_BF16(a0, b0, acc[0][0]);
      acc[0][1] = WMMA_BF16(a0, b1, acc[0][1]);
      acc[0][2] = WMMA_BF16(a0, b2, acc[0][2]);
      acc[0][3] = WMMA_BF16(a0, b3, acc[0][3]);
      acc[1][0] = WMMA_BF16(a1, b0, acc[1][0]);
      acc[1][1] = WMMA_BF16(a1, b1, acc[1][1]);
      acc[1][2] = WMMA_BF16(a1, b2, acc[1][2]);
      acc[1][3] = WMMA_BF16(a1, b3, acc[1][3]);
    }
    if (has_next) wait_async_all();
    __syncthreads();
  }

  const int n_lane = lane & 15;
  const int m_hi   = (lane >> 4) * 8;
#pragma unroll
  for (int mf = 0; mf < 2; ++mf) {
#pragma unroll
    for (int nf = 0; nf < 4; ++nf) {
#pragma unroll
      for (int r = 0; r < 8; ++r) {
        int m = rowBase + wm * 32 + mf * 16 + m_hi + r;
        int n = colBase + wn * 64 + nf * 16 + n_lane;
        if (m < M && n < N) {
          float v = acc[mf][nf][r];
          if (accum) {
            C[(size_t)m * ldc + n] += v;
          } else {
            if (bias0) v += bias0[n];
            if (bias1) v += bias1[n];
            C[(size_t)m * ldc + n] = v;
          }
        }
      }
    }
  }
}

// ---------------------------------------------------------------------------
// Softmax over S + context einsum ('sb,sbd->bd') -> bf16 xbuf[:, H:3H]
// ---------------------------------------------------------------------------
__global__ __launch_bounds__(256) void softmax_context_kernel(
    const float*   __restrict__ energies,  // [S*B]
    const uint8_t* __restrict__ mask,      // [S*B] bool
    const float*   __restrict__ enc,       // [S,B,2H]
    bf16_t*        __restrict__ xbuf)      // [B, 3H]
{
  __shared__ float w_s[Sdim];
  __shared__ float inv_s;
  int b = blockIdx.x;
  int t = threadIdx.x;
  if (t < Sdim) {
    float e = energies[t * Bdim + b];
    if (mask[t * Bdim + b]) e = -__builtin_inff();
    w_s[t] = e;
  }
  __syncthreads();
  if (t == 0) {
    float mx = -3.402823e38f;
    for (int s = 0; s < Sdim; ++s) mx = fmaxf(mx, w_s[s]);
    float sum = 0.f;
    for (int s = 0; s < Sdim; ++s) { float ev = fast_exp(w_s[s] - mx); w_s[s] = ev; sum += ev; }
    inv_s = 1.0f / sum;
  }
  __syncthreads();
  float acc[8] = {};
  for (int s = 0; s < Sdim; ++s) {
    float w = w_s[s];
    const float* row = enc + ((size_t)s * Bdim + b) * (2 * Hdim);
#pragma unroll
    for (int i = 0; i < 8; ++i) acc[i] += w * row[t + 256 * i];
  }
  float inv = inv_s;
#pragma unroll
  for (int i = 0; i < 8; ++i)
    xbuf[(size_t)b * (3 * Hdim) + Hdim + t + 256 * i] = (bf16_t)(acc[i] * inv);
}

// ---------------------------------------------------------------------------
// LSTM cell elementwise (gates order: i,f,g,o) — hardware sigmoid/tanh
// ---------------------------------------------------------------------------
__global__ __launch_bounds__(256) void lstm_cell_kernel(
    const float* __restrict__ gates,  // [B, 4H]
    const float* __restrict__ cell,   // [B, H]
    float* __restrict__ out_h, float* __restrict__ out_c,
    bf16_t* __restrict__ hnew_bf)
{
  int idx = blockIdx.x * blockDim.x + threadIdx.x;   // B*H
  int b = idx >> 10, h = idx & (Hdim - 1);
  const float* g = gates + (size_t)b * (4 * Hdim);
  float gi = g[h], gf = g[Hdim + h], gg = g[2 * Hdim + h], go = g[3 * Hdim + h];
  float c_new = fast_sigmoid(gf) * cell[idx] + fast_sigmoid(gi) * fast_tanh(gg);
  float h_new = fast_sigmoid(go) * fast_tanh(c_new);
  out_h[idx] = h_new;
  out_c[idx] = c_new;
  hnew_bf[idx] = (bf16_t)h_new;
}

// ---------------------------------------------------------------------------
// Row-wise log-softmax over VOC
// ---------------------------------------------------------------------------
__global__ __launch_bounds__(256) void log_softmax_kernel(
    const float* __restrict__ logits, float* __restrict__ out)
{
  __shared__ float red[256];
  int b = blockIdx.x, t = threadIdx.x;
  const float* row = logits + (size_t)b * VOCdim;
  float mx = -3.402823e38f;
  for (int j = t; j < VOCdim; j += 256) mx = fmaxf(mx, row[j]);
  red[t] = mx; __syncthreads();
  for (int s = 128; s > 0; s >>= 1) { if (t < s) red[t] = fmaxf(red[t], red[t + s]); __syncthreads(); }
  mx = red[0]; __syncthreads();
  float sum = 0.f;
  for (int j = t; j < VOCdim; j += 256) sum += fast_exp(row[j] - mx);
  red[t] = sum; __syncthreads();
  for (int s = 128; s > 0; s >>= 1) { if (t < s) red[t] += red[t + s]; __syncthreads(); }
  float lse = mx + fast_log(red[0]);
  for (int j = t; j < VOCdim; j += 256) out[(size_t)b * VOCdim + j] = row[j] - lse;
}

// ---------------------------------------------------------------------------
// Host orchestration
// ---------------------------------------------------------------------------
static inline size_t align256(size_t x) { return (x + 255) & ~(size_t)255; }

extern "C" void kernel_launch(void* const* d_in, const int* in_sizes, int n_in,
                              void* d_out, int out_size, void* d_ws, size_t ws_size,
                              hipStream_t stream) {
  const float*   hidden  = (const float*)d_in[0];     // [1,B,H]
  const float*   cell    = (const float*)d_in[1];     // [1,B,H]
  const float*   enc     = (const float*)d_in[2];     // [S,B,2H]
  const uint8_t* mask    = (const uint8_t*)d_in[3];   // [S,B] bool
  const float*   attn1_w = (const float*)d_in[4];     // [H,3H]
  const float*   attn1_b = (const float*)d_in[5];     // [H]
  const float*   attn2_w = (const float*)d_in[6];     // [1,H]
  // attn2_b (d_in[7]) cancels under softmax/log_softmax: intentionally unused
  const float*   w_ih    = (const float*)d_in[8];     // [4H,3H]
  const float*   w_hh    = (const float*)d_in[9];     // [4H,H]
  const float*   b_ih    = (const float*)d_in[10];    // [4H]
  const float*   b_hh    = (const float*)d_in[11];    // [4H]
  const float*   out_w   = (const float*)d_in[12];    // [VOC,H]
  const float*   out_b   = (const float*)d_in[13];    // [VOC]

  float* out = (float*)d_out;
  float* out_logp = out;                              // [B,VOC]
  float* out_h    = out + (size_t)Bdim * VOCdim;      // [B,H]
  float* out_c    = out_h + (size_t)Bdim * Hdim;      // [B,H]

  // Workspace layout
  char* w = (char*)d_ws;
  bf16_t* attn1w_bf = (bf16_t*)w; w += align256((size_t)Hdim * 3 * Hdim * 2);
  bf16_t* wih_bf    = (bf16_t*)w; w += align256((size_t)4 * Hdim * 3 * Hdim * 2);
  bf16_t* whh_bf    = (bf16_t*)w; w += align256((size_t)4 * Hdim * Hdim * 2);
  bf16_t* outw_bf   = (bf16_t*)w; w += align256((size_t)VOCdim * Hdim * 2);
  bf16_t* xbuf      = (bf16_t*)w; w += align256((size_t)Bdim * 3 * Hdim * 2);
  bf16_t* hnew_bf   = (bf16_t*)w; w += align256((size_t)Bdim * Hdim * 2);
  float*  hproj     = (float*)w;  w += align256((size_t)Bdim * Hdim * 4);
  float*  energies  = (float*)w;  w += align256((size_t)Sdim * Bdim * 4);
  float*  gates     = (float*)w;  w += align256((size_t)Bdim * 4 * Hdim * 4);
  float*  logits    = (float*)w;  w += align256((size_t)Bdim * VOCdim * 4);
  (void)ws_size; (void)n_in; (void)in_sizes; (void)out_size;

  // 1) weight conversions f32 -> bf16
  f32_to_bf16_kernel<<<4096, 256, 0, stream>>>(attn1_w, attn1w_bf, (size_t)Hdim * 3 * Hdim);
  f32_to_bf16_kernel<<<4096, 256, 0, stream>>>(w_ih,    wih_bf,    (size_t)4 * Hdim * 3 * Hdim);
  f32_to_bf16_kernel<<<4096, 256, 0, stream>>>(w_hh,    whh_bf,    (size_t)4 * Hdim * Hdim);
  f32_to_bf16_kernel<<<4096, 256, 0, stream>>>(out_w,   outw_bf,   (size_t)VOCdim * Hdim);
  pack_hidden_kernel<<<(Bdim * Hdim) / 256, 256, 0, stream>>>(hidden, xbuf);
  zero_kernel<<<64, 256, 0, stream>>>(energies, (size_t)Sdim * Bdim);

  // 2) hproj[B,H] = hidden @ attn1_w[:, :H].T
  gemm_bt_kernel<<<dim3(Bdim / BM, Hdim / BN), 256, 0, stream>>>(
      xbuf, 3 * Hdim, attn1w_bf, 3 * Hdim, hproj, Hdim,
      Bdim, Hdim, Hdim, nullptr, nullptr, 0);

  // 3) big attention GEMM + fused tanh/attn2 reduction -> energies
  attn_energy_kernel<<<dim3((Sdim * Bdim) / BM, Hdim / BN), 256, 0, stream>>>(
      enc, attn1w_bf + Hdim, hproj, attn1_b, attn2_w, energies);

  // 4) softmax over S + context -> xbuf[:, H:3H]
  softmax_context_kernel<<<Bdim, 256, 0, stream>>>(energies, mask, enc, xbuf);

  // 5) gates = x @ w_ih.T + b_ih + b_hh ; then += hidden @ w_hh.T
  gemm_bt_kernel<<<dim3(Bdim / BM, (4 * Hdim) / BN), 256, 0, stream>>>(
      xbuf, 3 * Hdim, wih_bf, 3 * Hdim, gates, 4 * Hdim,
      Bdim, 4 * Hdim, 3 * Hdim, b_ih, b_hh, 0);
  gemm_bt_kernel<<<dim3(Bdim / BM, (4 * Hdim) / BN), 256, 0, stream>>>(
      xbuf, 3 * Hdim, whh_bf, Hdim, gates, 4 * Hdim,
      Bdim, 4 * Hdim, Hdim, nullptr, nullptr, 1);

  // 6) LSTM elementwise -> h_new/c_new outputs + bf16 h_new
  lstm_cell_kernel<<<(Bdim * Hdim) / 256, 256, 0, stream>>>(
      gates, cell, out_h, out_c, hnew_bf);

  // 7) logits = h_new @ out_w.T + out_b ; log_softmax -> d_out
  gemm_bt_kernel<<<dim3(Bdim / BM, (VOCdim + BN - 1) / BN), 256, 0, stream>>>(
      hnew_bf, Hdim, outw_bf, Hdim, logits, VOCdim,
      Bdim, VOCdim, Hdim, out_b, nullptr, 0);
  log_softmax_kernel<<<Bdim, 256, 0, stream>>>(logits, out_logp);
}
